// Mamba2Triton_61512521614096
// MI455X (gfx1250) — matrix operand order repaired and checked by reading, not compiled, not run on hardware
//
#include <hip/hip_runtime.h>

// ---------------- CDNA5 WMMA types ----------------
typedef __attribute__((ext_vector_type(16))) __bf16       v16bf;
typedef __attribute__((ext_vector_type(8)))  float        v8f;
typedef __attribute__((ext_vector_type(4)))  unsigned int v4u;
typedef __attribute__((ext_vector_type(2)))  unsigned int v2u;

#define BATCH    2
#define SEQLEN   2048
#define NHEADS   32
#define HEADDIM  64
#define DSTATE   128
#define CHUNK    256
#define NCHUNK   (SEQLEN / CHUNK)   // 8

// padded LDS row strides (elements). 136*2/4 = 68 = 4 mod 64 banks,
// 264*2/4 = 132 = 4 mod 64 banks -> 16 lanes * 4 banks = all 64 banks.
#define RSN 136   // rows of 128 (+8 pad)
#define RSL 264   // rows of 256 (+8 pad)

__device__ __forceinline__ v8f wmma_bf16(v16bf a, v16bf b, v8f c) {
  return __builtin_amdgcn_wmma_f32_16x16x32_bf16(false, a, false, b, (short)0, c, false, false);
}

union V16 { v16bf bf; v4u u[2]; };
union BF4 { __bf16 b[4]; v2u u; };

// A-operand (16xK slab): lane m = lo; elements i <-> K = k0+16*hiL+i.
// Two ds_load_b128.
__device__ __forceinline__ v16bf load_a_op(const __bf16* row, int k0, int hiL) {
  V16 t;
  t.u[0] = *(const v4u*)(row + k0 + 16 * hiL);
  t.u[1] = *(const v4u*)(row + k0 + 16 * hiL + 8);
  return t.bf;
}
// B-operand (Kx16 slab): lane n = lo; elements i <-> K = k0+16*hiL+i. Two ds_load_b128.
__device__ __forceinline__ v16bf load_b_op(const __bf16* row, int k0, int hiL) {
  V16 t;
  const __bf16* p = row + k0 + 16 * hiL;
  t.u[0] = *(const v4u*)(p);
  t.u[1] = *(const v4u*)(p + 8);
  return t.bf;
}
__device__ __forceinline__ void store_bf4(__bf16* dst, float4 v, float s) {
  BF4 t;
  t.b[0] = (__bf16)(v.x * s); t.b[1] = (__bf16)(v.y * s);
  t.b[2] = (__bf16)(v.z * s); t.b[3] = (__bf16)(v.w * s);
  *(v2u*)dst = t.u;
}

// ---------------- Kernel 1: per-chunk states ----------------
// states[b,c,h,p,n] = sum_l B[l,n]*exp(Atot-Acum[l])*(x[l,p]*dt[l])
// GEMM M=p(64) N=n(128) K=l(256). Staging transposed: K=l innermost.
#define S1_XDT 0                                   // [64][RSL]
#define S1_BT  (64 * RSL * 2)                      // [128][RSL]
#define S1_AC  (S1_BT + 128 * RSL * 2)
#define S1_TOTAL (S1_AC + CHUNK * 4)

__global__ void __launch_bounds__(256)
k_states(const float* __restrict__ x, const float* __restrict__ dt,
         const float* __restrict__ A, const float* __restrict__ Bg,
         float* __restrict__ states, float* __restrict__ chunk_sum) {
  extern __shared__ char smem[];
  __bf16* sXdT = (__bf16*)(smem + S1_XDT);
  __bf16* sBT  = (__bf16*)(smem + S1_BT);
  float*  sAc  = (float*)(smem + S1_AC);

  const int hi = blockIdx.x, ci = blockIdx.y, bi = blockIdx.z;
  const int t  = threadIdx.x;                 // l within chunk
  const size_t gl = (size_t)bi * SEQLEN + (size_t)ci * CHUNK + t;

  const float dtv = dt[gl * NHEADS + hi];
  sAc[t] = A[hi] * dtv;
  __syncthreads();
  for (int off = 1; off < CHUNK; off <<= 1) {
    float v = (t >= off) ? sAc[t - off] : 0.f;
    __syncthreads();
    sAc[t] += v;
    __syncthreads();
  }
  const float ctot  = sAc[CHUNK - 1];
  const float scale = dtv * __expf(ctot - sAc[t]);

  const float4* x4 = (const float4*)(x + (gl * NHEADS + hi) * HEADDIM);
  #pragma unroll
  for (int q = 0; q < HEADDIM / 4; ++q) {      // transpose scatter: sXdT[p][t]
    float4 v = x4[q];
    sXdT[(q*4+0)*RSL + t] = (__bf16)(v.x * scale);
    sXdT[(q*4+1)*RSL + t] = (__bf16)(v.y * scale);
    sXdT[(q*4+2)*RSL + t] = (__bf16)(v.z * scale);
    sXdT[(q*4+3)*RSL + t] = (__bf16)(v.w * scale);
  }
  const float4* b4 = (const float4*)(Bg + gl * DSTATE);
  #pragma unroll
  for (int q = 0; q < DSTATE / 4; ++q) {       // transpose scatter: sBT[n][t]
    float4 v = b4[q];
    sBT[(q*4+0)*RSL + t] = (__bf16)v.x;
    sBT[(q*4+1)*RSL + t] = (__bf16)v.y;
    sBT[(q*4+2)*RSL + t] = (__bf16)v.z;
    sBT[(q*4+3)*RSL + t] = (__bf16)v.w;
  }
  if (t == 0) chunk_sum[((size_t)bi * NHEADS + hi) * NCHUNK + ci] = ctot;
  __syncthreads();

  const int w = t >> 5, lane = t & 31, lo = lane & 15, hiL = lane >> 4;
  const int p0 = (w >> 1) * 16;                // wave's p-tile (shared A-op)
  const int n0b = (w & 1) * 64;                // wave's 4 n-tiles
  const __bf16* arow = sXdT + (size_t)(p0 + lo) * RSL;

  v8f acc[4] = {{}, {}, {}, {}};
  for (int kk = 0; kk < CHUNK / 32; ++kk) {
    const int k0 = kk * 32;
    v16bf a = load_a_op(arow, k0, hiL);        // A(m=p, k=l)
    #pragma unroll
    for (int j = 0; j < 4; ++j) {
      v16bf b = load_b_op(sBT + (size_t)(n0b + j*16 + lo) * RSL, k0, hiL); // B(k=l,n)
      acc[j] = wmma_bf16(a, b, acc[j]);
    }
  }
  float* sbase = states + (((size_t)(bi * NCHUNK + ci) * NHEADS + hi) * HEADDIM) * DSTATE;
  #pragma unroll
  for (int j = 0; j < 4; ++j)
    #pragma unroll
    for (int r = 0; r < 8; ++r)
      sbase[(size_t)(p0 + r + 8*hiL) * DSTATE + n0b + j*16 + lo] = acc[j][r];
}

// ---------------- Kernel 2: inter-chunk recurrence ----------------
__global__ void k_scan(const float* __restrict__ states, const float* __restrict__ chunk_sum,
                       float* __restrict__ prev) {
  const int bh = blockIdx.x;
  const int bi = bh / NHEADS, hi = bh % NHEADS;
  float cs[NCHUNK];
  #pragma unroll
  for (int ci = 0; ci < NCHUNK; ++ci)
    cs[ci] = __expf(chunk_sum[(size_t)bh * NCHUNK + ci]);

  const int PN = HEADDIM * DSTATE;
  for (int e = threadIdx.x; e < PN; e += blockDim.x) {
    float run = 0.f;
    #pragma unroll
    for (int ci = 0; ci < NCHUNK; ++ci) {
      size_t idx = ((size_t)(bi * NCHUNK + ci) * NHEADS + hi) * PN + e;
      prev[idx] = run;
      run = run * cs[ci] + states[idx];
    }
  }
}

// ---------------- Kernel 3: Y = (C B^T ∘ L) Xd + diag(exp(Acum)) (C prev^T) ----------------
#define S3_C   0                                   // [256][RSN]  K=n innermost
#define S3_B   (256 * RSN * 2)                     // [256][RSN]  K=n innermost
#define S3_XT  (S3_B + 256 * RSN * 2)              // [64][RSL]   K=s innermost
#define S3_P   (S3_XT + 64 * RSL * 2)              // [64][RSN]   K=n innermost
#define S3_G   (S3_P + 64 * RSN * 2)               // [8*16][RSL] per-wave, K=s innermost
#define S3_AC  (S3_G + 128 * RSL * 2)
#define S3_TOTAL (S3_AC + CHUNK * 4)               // 259072 B (< 320KB/WGP)

__global__ void __launch_bounds__(256)
k_out(const float* __restrict__ x, const float* __restrict__ dt,
      const float* __restrict__ A, const float* __restrict__ Bg,
      const float* __restrict__ Cg, const float* __restrict__ prev,
      float* __restrict__ out) {
  extern __shared__ char smem[];
  __bf16* sC  = (__bf16*)(smem + S3_C);
  __bf16* sB  = (__bf16*)(smem + S3_B);
  __bf16* sXT = (__bf16*)(smem + S3_XT);
  __bf16* sP  = (__bf16*)(smem + S3_P);
  __bf16* sG  = (__bf16*)(smem + S3_G);
  float*  sAc = (float*)(smem + S3_AC);

  const int hi = blockIdx.x, ci = blockIdx.y, bi = blockIdx.z;
  const int t  = threadIdx.x;
  const size_t gl = (size_t)bi * SEQLEN + (size_t)ci * CHUNK + t;

  const float dtv = dt[gl * NHEADS + hi];
  sAc[t] = A[hi] * dtv;
  __syncthreads();
  for (int off = 1; off < CHUNK; off <<= 1) {
    float v = (t >= off) ? sAc[t - off] : 0.f;
    __syncthreads();
    sAc[t] += v;
    __syncthreads();
  }
  const float4* x4 = (const float4*)(x + (gl * NHEADS + hi) * HEADDIM);
  #pragma unroll
  for (int q = 0; q < HEADDIM / 4; ++q) {      // transpose scatter: sXT[p][t] (Xd, no decay)
    float4 v = x4[q];
    sXT[(q*4+0)*RSL + t] = (__bf16)(v.x * dtv);
    sXT[(q*4+1)*RSL + t] = (__bf16)(v.y * dtv);
    sXT[(q*4+2)*RSL + t] = (__bf16)(v.z * dtv);
    sXT[(q*4+3)*RSL + t] = (__bf16)(v.w * dtv);
  }
  const float4* b4 = (const float4*)(Bg + gl * DSTATE);
  const float4* c4 = (const float4*)(Cg + gl * DSTATE);
  #pragma unroll
  for (int q = 0; q < DSTATE / 4; ++q) {       // row-major, packed 8B stores
    store_bf4(sB + (size_t)t * RSN + q*4, b4[q], 1.f);
    store_bf4(sC + (size_t)t * RSN + q*4, c4[q], 1.f);
  }
  const float4* p4 = (const float4*)(prev + ((size_t)(bi * NCHUNK + ci) * NHEADS + hi)
                                            * HEADDIM * DSTATE);
  #pragma unroll
  for (int j = 0; j < (HEADDIM * DSTATE / 4) / 256; ++j) {   // 8 iters
    int idx4 = t + 256 * j;
    int p = idx4 >> 5, n = (idx4 & 31) * 4;
    store_bf4(sP + (size_t)p * RSN + n, p4[idx4], 1.f);
  }
  __syncthreads();

  const int w = t >> 5, lane = t & 31, lo = lane & 15, hiL = lane >> 4;

  for (int li = 0; li < 2; ++li) {
    const int lt = (li == 0) ? w : 15 - w;     // balanced pairing: (w, 15-w)
    const int l0 = lt * 16;

    // shared A-operand for C-row GEMMs (reused by G-build and off-part)
    v16bf aC[4];
    #pragma unroll
    for (int kk = 0; kk < 4; ++kk)
      aC[kk] = load_a_op(sC + (size_t)(l0 + lo) * RSN, kk * 32, hiL);

    // --- G = C B^T, lower-triangular s-tiles only, mask with exp(segsum) ---
    for (int st = 0; st <= lt; ++st) {
      const int s0 = st * 16;
      v8f g = {};
      #pragma unroll
      for (int kk = 0; kk < 4; ++kk) {
        v16bf b = load_b_op(sB + (size_t)(s0 + lo) * RSN, kk * 32, hiL); // B(k=n,N=s)
        g = wmma_bf16(aC[kk], b, g);
      }
      #pragma unroll
      for (int r = 0; r < 8; ++r) {
        int lrow = r + 8 * hiL;
        int l = l0 + lrow, s = s0 + lo;
        float v = (s <= l) ? g[r] * __expf(sAc[l] - sAc[s]) : 0.f;
        sG[(size_t)(w * 16 + lrow) * RSL + s] = (__bf16)v;
      }
    }
    if (!(lt & 1)) {                           // zero-pad to a full K=32 step
      int s0 = (lt + 1) * 16;
      #pragma unroll
      for (int r = 0; r < 8; ++r)
        sG[(size_t)(w * 16 + r + 8*hiL) * RSL + s0 + lo] = (__bf16)0.f;
    }

    const int nk = (lt + 2) >> 1;
    float esc[8];
    #pragma unroll
    for (int r = 0; r < 8; ++r) esc[r] = __expf(sAc[l0 + r + 8*hiL]); // out_decay rows

    v8f acc[4]  = {{}, {}, {}, {}};            // Y_diag, 4 p-tiles
    v8f acc2[4] = {{}, {}, {}, {}};            // C prev^T (unscaled)
    for (int kk = 0; kk < nk; ++kk) {
      const int k0 = kk * 32;
      v16bf a = load_a_op(sG + (size_t)(w * 16 + lo) * RSL, k0, hiL);  // A(m=l,k=s)
      #pragma unroll
      for (int pt = 0; pt < 4; ++pt) {
        v16bf b = load_b_op(sXT + (size_t)(pt*16 + lo) * RSL, k0, hiL); // B(k=s,N=p)
        acc[pt] = wmma_bf16(a, b, acc[pt]);
      }
    }
    #pragma unroll
    for (int kk = 0; kk < 4; ++kk) {
      const int k0 = kk * 32;
      #pragma unroll
      for (int pt = 0; pt < 4; ++pt) {
        v16bf b = load_b_op(sP + (size_t)(pt*16 + lo) * RSN, k0, hiL);  // B(k=n,N=p)
        acc2[pt] = wmma_bf16(aC[kk], b, acc2[pt]);
      }
    }
    #pragma unroll
    for (int pt = 0; pt < 4; ++pt)
      #pragma unroll
      for (int r = 0; r < 8; ++r) {
        int l = l0 + r + 8 * hiL;
        out[(((size_t)bi * SEQLEN + (size_t)ci * CHUNK + l) * NHEADS + hi) * HEADDIM
            + pt*16 + lo] = acc[pt][r] + esc[r] * acc2[pt][r];
      }
  }
}

// ---------------- Host launch ----------------
extern "C" void kernel_launch(void* const* d_in, const int* in_sizes, int n_in,
                              void* d_out, int out_size, void* d_ws, size_t ws_size,
                              hipStream_t stream) {
  (void)in_sizes; (void)n_in; (void)out_size; (void)ws_size;
  const float* x  = (const float*)d_in[0];
  const float* dt = (const float*)d_in[1];
  const float* A  = (const float*)d_in[2];
  const float* Bg = (const float*)d_in[3];
  const float* Cg = (const float*)d_in[4];
  float* out = (float*)d_out;

  const size_t STATES_ELEMS = (size_t)BATCH * NCHUNK * NHEADS * HEADDIM * DSTATE;
  char* ws = (char*)d_ws;
  float* states = (float*)ws;
  float* prevs  = (float*)(ws + STATES_ELEMS * sizeof(float));
  float* csum   = (float*)(ws + 2 * STATES_ELEMS * sizeof(float));

  hipFuncSetAttribute((const void*)k_states, hipFuncAttributeMaxDynamicSharedMemorySize, S1_TOTAL);
  hipFuncSetAttribute((const void*)k_out,    hipFuncAttributeMaxDynamicSharedMemorySize, S3_TOTAL);

  dim3 grid(NHEADS, NCHUNK, BATCH);
  k_states<<<grid, 256, S1_TOTAL, stream>>>(x, dt, A, Bg, states, csum);
  k_scan<<<BATCH * NHEADS, 256, 0, stream>>>(states, csum, prevs);
  k_out<<<grid, 256, S3_TOTAL, stream>>>(x, dt, A, Bg, Cg, prevs, out);
}
